// DONNSentimentClassifier_76862734729917
// MI455X (gfx1250) — matrix operand order repaired and checked by reading, not compiled, not run on hardware
//
#include <hip/hip_runtime.h>
#include <hip/hip_bf16.h>
#include <math.h>
#include <stdint.h>

// ---------------- problem constants (match reference) ----------------
#define BB   256
#define TT   512
#define UU   64
#define EDIM 100
#define PDIM 20
#define NCLS 2
#define ROWS (BB * TT)          // 131072 flattened (b,t) rows
#define TILES (ROWS / 16)       // 8192 row-tiles of 16
#define KPAD 128                // embedding K padded to WMMA multiple

#define HOPF_SCALE 0.2f
#define HOPF_DT    1e-3f

typedef __bf16 bf16_t;
typedef bf16_t v16bf __attribute__((ext_vector_type(16)));
typedef bf16_t v8bf  __attribute__((ext_vector_type(8)));
typedef float  v8f   __attribute__((ext_vector_type(8)));

// =====================================================================
// Fragment helpers (wave32, v_wmma_f32_16x16x32_bf16)
// =====================================================================
// A: 16xK bf16 tile in LDS, row-major, leading dim ld (elements).
// ISA A layout: lanes 0-15 -> M=lane, khalf 0; lanes 16-31 -> M=lane-16,
// khalf 8. VGPR pairs j=0..3 hold K k0+khalf+0..7, j=4..7 hold
// K k0+16+khalf+0..7 -> two contiguous 16-byte LDS loads per fragment.
static __device__ __attribute__((always_inline)) v16bf
load_a_frag(const bf16_t* tile, int ld, int k0) {
  const int lane = threadIdx.x & 31;
  const int m = lane & 15;
  const int khalf = (lane & 16) ? 8 : 0;
  const bf16_t* base = tile + m * ld + k0 + khalf;
  v8bf lo = *(const v8bf*)(base);
  v8bf hi = *(const v8bf*)(base + 16);
  return __builtin_shufflevector(lo, hi, 0, 1, 2, 3, 4, 5, 6, 7,
                                 8, 9, 10, 11, 12, 13, 14, 15);
}

// B: pre-packed bf16 fragments in global memory:
//    P[((nt*nkt + kt)*32 + lane)*16 .. +15]  (one 32B load per lane)
static __device__ __attribute__((always_inline)) v16bf
load_b_frag(const bf16_t* __restrict__ P, int nkt, int nt, int kt) {
  const int lane = threadIdx.x & 31;
  return *(const v16bf*)(P + (((size_t)(nt * nkt + kt) * 32 + lane) << 4));
}

static __device__ __attribute__((always_inline)) v8f
wmma_bf16(v16bf a, v16bf b, v8f c) {
  return __builtin_amdgcn_wmma_f32_16x16x32_bf16(
      /*neg_a=*/false, a, /*neg_b=*/false, b,
      /*c_mod=*/(short)0, c, /*reuse_a=*/false, /*reuse_b=*/false);
}

static __device__ __attribute__((always_inline)) v8f zero8() {
  v8f z = {0.f, 0.f, 0.f, 0.f, 0.f, 0.f, 0.f, 0.f};
  return z;
}

// ---------------------------------------------------------------------
// Async-stage a 16x128 bf16 [zr | zi] tile into LDS using
// GLOBAL_LOAD_ASYNC_TO_LDS_B128 (ASYNCcnt path, no VGPR data movement).
// LDS rows are 256B: low 128B from zr, high 128B from zi.
// ---------------------------------------------------------------------
static __device__ __attribute__((always_inline)) void
async_stage_z(bf16_t* Z, const bf16_t* zr, const bf16_t* zi, int row0) {
  const unsigned lane = threadIdx.x & 31;
  const unsigned zlds = (unsigned)(uintptr_t)(void*)Z;  // in-LDS byte offset
  const unsigned lane16 = lane << 4;                     // 16B per lane
  const unsigned long long gr = (unsigned long long)(uintptr_t)zr;
  const unsigned long long gi = (unsigned long long)(uintptr_t)zi;
#pragma unroll
  for (int it = 0; it < 4; ++it) {
    const unsigned byteoff = (unsigned)it * 512u + lane16;  // 0..2047
    const unsigned r = byteoff >> 7;                        // 128B per src row
    const unsigned cb = byteoff & 127u;
    const unsigned long long srow = ((unsigned long long)(row0 + (int)r) * UU) * 2ull + cb;
    const unsigned ldsa_r = zlds + r * 256u + cb;
    const unsigned ldsa_i = ldsa_r + 128u;
    const unsigned long long gar = gr + srow;
    const unsigned long long gai = gi + srow;
    asm volatile("global_load_async_to_lds_b128 %0, %1, off"
                 :: "v"(ldsa_r), "v"(gar) : "memory");
    asm volatile("global_load_async_to_lds_b128 %0, %1, off"
                 :: "v"(ldsa_i), "v"(gai) : "memory");
  }
  asm volatile("s_wait_asynccnt 0x0" ::: "memory");
}

// =====================================================================
// Kernel 0: pack one KxN f32 weight matrix into bf16 B-fragment layout,
// zero-padded to Kp (multiple of 32). Runs once per weight, tiny.
// =====================================================================
__global__ __launch_bounds__(64) void k_pack_w(
    const float* __restrict__ W, int K, int Kp, int N,
    bf16_t* __restrict__ out) {
  const int nkt = Kp / 32;
  const int total = (N / 16) * nkt * 32;   // one thread per (nt,kt,lane)
  const int tid = blockIdx.x * blockDim.x + threadIdx.x;
  if (tid >= total) return;
  const int lane = tid & 31;
  const int kt = (tid >> 5) % nkt;
  const int nt = tid / (32 * nkt);
  const int n = nt * 16 + (lane & 15);
  const int kh = (lane & 16) ? 8 : 0;
  bf16_t* dst = out + (size_t)tid * 16;
#pragma unroll
  for (int j = 0; j < 8; ++j) {
    const int k = kt * 32 + kh + 2 * j + ((j & 4) ? 8 : 0);
    dst[2 * j]     = (k     < K) ? (bf16_t)W[(size_t)k       * N + n] : (bf16_t)0.f;
    dst[2 * j + 1] = (k + 1 < K) ? (bf16_t)W[(size_t)(k + 1) * N + n] : (bf16_t)0.f;
  }
}

// =====================================================================
// Kernel 1: embedding gather + first complex projection (bf16 out)
// =====================================================================
__global__ __launch_bounds__(32) void k_embed_proj1(
    const int* __restrict__ x, const float* __restrict__ E,
    const bf16_t* __restrict__ PW1r, const float* __restrict__ b1r,
    const bf16_t* __restrict__ PW1i, const float* __restrict__ b1i,
    bf16_t* __restrict__ x1r, bf16_t* __restrict__ x1i) {
  __shared__ bf16_t A[16 * KPAD];
  __shared__ int Tok[16];
  const int row0 = blockIdx.x * 16;
  const int lane = threadIdx.x;

  if (lane < 16) Tok[lane] = x[row0 + lane];
  // zero the K padding region (100..127 of each row)
  for (int idx = lane; idx < 16 * KPAD; idx += 32) A[idx] = (bf16_t)0.f;
  __syncthreads();
  // gather + convert embedding rows (cvt once per element, not per use)
  for (int idx = lane; idx < 16 * EDIM; idx += 32) {
    const int r = idx / EDIM;
    const int k = idx - r * EDIM;
    A[r * KPAD + k] = (bf16_t)E[(size_t)Tok[r] * EDIM + k];
  }
  __syncthreads();

#pragma unroll
  for (int half = 0; half < 2; ++half) {
    const bf16_t* P    = half ? PW1i : PW1r;
    const float*  bias = half ? b1i : b1r;
    bf16_t*       out  = half ? x1i : x1r;
#pragma unroll
    for (int nt = 0; nt < 4; ++nt) {
      v8f acc = zero8();
#pragma unroll
      for (int kt = 0; kt < 4; ++kt)
        acc = wmma_bf16(load_a_frag(A, KPAD, kt * 32),
                        load_b_frag(P, 4, nt, kt), acc);
      const int col = nt * 16 + (lane & 15);
      const float bv = bias[col];
      const int mbase = (lane & 16) ? 8 : 0;
#pragma unroll
      for (int g = 0; g < 8; ++g) {
        const int m = mbase + g;
        out[(size_t)(row0 + m) * UU + col] = (bf16_t)fmaxf(acc[g] + bv, 0.f);
      }
    }
  }
}

// =====================================================================
// Kernel 2: Hopf oscillator scan. One thread per (b,u); f32 state in
// registers across the whole T=512 scan, 20 unrolled fmaf substeps.
// =====================================================================
__global__ __launch_bounds__(256) void k_hopf(
    const bf16_t* __restrict__ xr, const bf16_t* __restrict__ xi,
    const float* __restrict__ om,
    bf16_t* __restrict__ zr_out, bf16_t* __restrict__ zi_out) {
  const int gid = blockIdx.x * blockDim.x + threadIdx.x;  // 0 .. B*U-1
  const int b = gid / UU;
  const int u = gid - b * UU;
  const float w = om[u];
  float zr = 0.1f, zi = 0.f;
  const size_t base = (size_t)b * TT * UU + u;
  for (int t = 0; t < TT; ++t) {
    const size_t off = base + (size_t)t * UU;
    const float fr = HOPF_SCALE * (float)xr[off];
    const float fi = HOPF_SCALE * (float)xi[off];
#pragma unroll
    for (int s = 0; s < 20; ++s) {
      const float r2  = fmaf(zr, zr, zi * zi);
      const float g   = 1.0f - r2;  // MU = 1
      const float dzr = fmaf(g, zr, fmaf(-w, zi, fr));
      const float dzi = fmaf(g, zi, fmaf(w, zr, fi));
      zr = fmaf(HOPF_DT, dzr, zr);
      zi = fmaf(HOPF_DT, dzi, zi);
    }
    zr_out[off] = (bf16_t)zr;
    zi_out[off] = (bf16_t)zi;
  }
}

// =====================================================================
// Kernel 3: h1 = relu([z1r|z1i] @ Wp1 + bp1); x2{r,i} = relu(h1 @ W2 + b2)
// =====================================================================
__global__ __launch_bounds__(32) void k_mid(
    const bf16_t* __restrict__ z1r, const bf16_t* __restrict__ z1i,
    const bf16_t* __restrict__ PWp1, const float* __restrict__ bp1,
    const bf16_t* __restrict__ PW2r, const float* __restrict__ b2r,
    const bf16_t* __restrict__ PW2i, const float* __restrict__ b2i,
    bf16_t* __restrict__ x2r, bf16_t* __restrict__ x2i) {
  __shared__ bf16_t Z[16 * 128];
  __shared__ bf16_t H1[16 * 64];
  const int row0 = blockIdx.x * 16;
  const int lane = threadIdx.x;

  // async DMA [z1r | z1i] tile straight into LDS
  async_stage_z(Z, z1r, z1i, row0);
  __syncthreads();

  // h1 (K=128) -> LDS as bf16
#pragma unroll
  for (int nt = 0; nt < 4; ++nt) {
    v8f acc = zero8();
#pragma unroll
    for (int kt = 0; kt < 4; ++kt)
      acc = wmma_bf16(load_a_frag(Z, 128, kt * 32),
                      load_b_frag(PWp1, 4, nt, kt), acc);
    const int col = nt * 16 + (lane & 15);
    const float bv = bp1[col];
    const int mbase = (lane & 16) ? 8 : 0;
#pragma unroll
    for (int g = 0; g < 8; ++g)
      H1[(mbase + g) * 64 + col] = (bf16_t)fmaxf(acc[g] + bv, 0.f);
  }
  __syncthreads();

  // x2r / x2i (K=64)
#pragma unroll
  for (int half = 0; half < 2; ++half) {
    const bf16_t* P    = half ? PW2i : PW2r;
    const float*  bias = half ? b2i : b2r;
    bf16_t*       out  = half ? x2i : x2r;
#pragma unroll
    for (int nt = 0; nt < 4; ++nt) {
      v8f acc = zero8();
#pragma unroll
      for (int kt = 0; kt < 2; ++kt)
        acc = wmma_bf16(load_a_frag(H1, 64, kt * 32),
                        load_b_frag(P, 2, nt, kt), acc);
      const int col = nt * 16 + (lane & 15);
      const float bv = bias[col];
      const int mbase = (lane & 16) ? 8 : 0;
#pragma unroll
      for (int g = 0; g < 8; ++g) {
        const int m = mbase + g;
        out[(size_t)(row0 + m) * UU + col] = (bf16_t)fmaxf(acc[g] + bv, 0.f);
      }
    }
  }
}

// =====================================================================
// Kernel 5: h2 (WMMA) -> tanh head (64x20) -> logits (20x2)
// =====================================================================
__global__ __launch_bounds__(32) void k_out(
    const bf16_t* __restrict__ z2r, const bf16_t* __restrict__ z2i,
    const bf16_t* __restrict__ PWp2, const float* __restrict__ bp2,
    const float* __restrict__ Wpr, const float* __restrict__ bpr,
    const float* __restrict__ Wh,  const float* __restrict__ bh,
    float* __restrict__ out) {
  __shared__ bf16_t Z[16 * 128];
  __shared__ float H2[16 * 64];
  __shared__ float H3[16 * PDIM];
  const int row0 = blockIdx.x * 16;
  const int lane = threadIdx.x;

  async_stage_z(Z, z2r, z2i, row0);
  __syncthreads();

  // h2 = relu([z2r|z2i] @ Wp2 + bp2) -> f32 LDS (feeds scalar head)
#pragma unroll
  for (int nt = 0; nt < 4; ++nt) {
    v8f acc = zero8();
#pragma unroll
    for (int kt = 0; kt < 4; ++kt)
      acc = wmma_bf16(load_a_frag(Z, 128, kt * 32),
                      load_b_frag(PWp2, 4, nt, kt), acc);
    const int col = nt * 16 + (lane & 15);
    const float bv = bp2[col];
    const int mbase = (lane & 16) ? 8 : 0;
#pragma unroll
    for (int g = 0; g < 8; ++g)
      H2[(mbase + g) * 64 + col] = fmaxf(acc[g] + bv, 0.f);
  }
  __syncthreads();

  // h3 = tanh(h2 @ Wpr + bpr): 20 outputs split across lane pairs
  {
    const int r  = lane & 15;
    const int p0 = (lane & 16) ? (PDIM / 2) : 0;
    const float* h2row = &H2[r * 64];
#pragma unroll
    for (int p = p0; p < p0 + PDIM / 2; ++p) {
      float s = bpr[p];
#pragma unroll
      for (int k = 0; k < UU; ++k)
        s = fmaf(h2row[k], Wpr[(size_t)k * PDIM + p], s);
      H3[r * PDIM + p] = tanhf(s);
    }
  }
  __syncthreads();

  // logits = h3 @ Wh + bh  (f32 output)
  if (lane < 16) {
    const int r = lane;
#pragma unroll
    for (int c = 0; c < NCLS; ++c) {
      float s = bh[c];
#pragma unroll
      for (int p = 0; p < PDIM; ++p)
        s = fmaf(H3[r * PDIM + p], Wh[p * NCLS + c], s);
      out[(size_t)(row0 + r) * NCLS + c] = s;
    }
  }
}

// =====================================================================
// host-side launch
// =====================================================================
extern "C" void kernel_launch(void* const* d_in, const int* in_sizes, int n_in,
                              void* d_out, int out_size, void* d_ws, size_t ws_size,
                              hipStream_t stream) {
  (void)in_sizes; (void)n_in; (void)out_size; (void)ws_size;

  const int*   x   = (const int*)  d_in[0];
  const float* E   = (const float*)d_in[1];
  const float* W1r = (const float*)d_in[2];
  const float* b1r = (const float*)d_in[3];
  const float* W1i = (const float*)d_in[4];
  const float* b1i = (const float*)d_in[5];
  const float* om1 = (const float*)d_in[6];
  const float* Wp1 = (const float*)d_in[7];
  const float* bp1 = (const float*)d_in[8];
  const float* W2r = (const float*)d_in[9];
  const float* b2r = (const float*)d_in[10];
  const float* W2i = (const float*)d_in[11];
  const float* b2i = (const float*)d_in[12];
  const float* om2 = (const float*)d_in[13];
  const float* Wp2 = (const float*)d_in[14];
  const float* bp2 = (const float*)d_in[15];
  const float* Wpr = (const float*)d_in[16];
  const float* bpr = (const float*)d_in[17];
  const float* Wh  = (const float*)d_in[18];
  const float* bh  = (const float*)d_in[19];
  float* out = (float*)d_out;

  const size_t BTU = (size_t)BB * TT * UU;  // 8,388,608 elements

  // bf16 workspace layout: 4 activation ping-pong buffers + packed weights
  bf16_t* wsb = (bf16_t*)d_ws;
  bf16_t* bufXr = wsb;             // x-projection real (layer 1 & 2)
  bf16_t* bufXi = wsb + BTU;       // x-projection imag
  bf16_t* bufZr = wsb + 2 * BTU;   // oscillator state real
  bf16_t* bufZi = wsb + 3 * BTU;   // oscillator state imag
  bf16_t* p = wsb + 4 * BTU;
  bf16_t* PW1r = p; p += (size_t)KPAD * UU;   // 128x64
  bf16_t* PW1i = p; p += (size_t)KPAD * UU;
  bf16_t* PWp1 = p; p += (size_t)128 * UU;
  bf16_t* PW2r = p; p += (size_t)64 * UU;
  bf16_t* PW2i = p; p += (size_t)64 * UU;
  bf16_t* PWp2 = p; p += (size_t)128 * UU;

  // 0) pre-pack weights into bf16 fragment layout (tiny, once per call)
  k_pack_w<<<8, 64, 0, stream>>>(W1r, EDIM, KPAD, UU, PW1r);
  k_pack_w<<<8, 64, 0, stream>>>(W1i, EDIM, KPAD, UU, PW1i);
  k_pack_w<<<8, 64, 0, stream>>>(Wp1, 128, 128, UU, PWp1);
  k_pack_w<<<4, 64, 0, stream>>>(W2r, 64, 64, UU, PW2r);
  k_pack_w<<<4, 64, 0, stream>>>(W2i, 64, 64, UU, PW2i);
  k_pack_w<<<8, 64, 0, stream>>>(Wp2, 128, 128, UU, PWp2);

  // 1) embedding + first complex projection
  k_embed_proj1<<<TILES, 32, 0, stream>>>(x, E, PW1r, b1r, PW1i, b1i,
                                          bufXr, bufXi);
  // 2) Hopf layer 1
  k_hopf<<<(BB * UU) / 256, 256, 0, stream>>>(bufXr, bufXi, om1, bufZr, bufZi);
  // 3) h1 projection + second complex projection
  k_mid<<<TILES, 32, 0, stream>>>(bufZr, bufZi, PWp1, bp1,
                                  PW2r, b2r, PW2i, b2i, bufXr, bufXi);
  // 4) Hopf layer 2
  k_hopf<<<(BB * UU) / 256, 256, 0, stream>>>(bufXr, bufXi, om2, bufZr, bufZi);
  // 5) h2 projection + tanh head + logits
  k_out<<<TILES, 32, 0, stream>>>(bufZr, bufZi, PWp2, bp2,
                                  Wpr, bpr, Wh, bh, out);
}
